// NystromAttention_75986561401390
// MI455X (gfx1250) — compile-verified, matrix operand order checked by reading, not run on hardware
//
#include <hip/hip_runtime.h>

// ---------------------------------------------------------------------------
// Nystrom attention for MI455X (gfx1250, wave32, WMMA 16x16x32 f16 -> f32).
// All GEMMs run on v_wmma_f32_16x16x32_f16. Every B operand is consumed as a
// row-major "BT" (K contiguous), produced either by a transpose-prep kernel
// (weights) or by the producing GEMM's dual out/outT epilogue.
// ---------------------------------------------------------------------------

typedef __attribute__((ext_vector_type(16))) _Float16 v16h;
typedef __attribute__((ext_vector_type(8)))  _Float16 v8h;
typedef __attribute__((ext_vector_type(8)))  float    v8f;

#define B_    4
#define N_    4096
#define DIM_  512
#define H_    8
#define DH_   64
#define M_    256
#define L_    16          // tokens per landmark group
#define BH_   (B_ * H_)   // 32
#define NN_   (B_ * N_)   // 16384
#define QKVN_ (3 * DIM_)  // 1536

// ---- WMMA fragment loader -------------------------------------------------
// 16-bit A-matrix 16x32 layout (ISA 7.12.2): lane<16 holds row r=lane, K in
// {0..7} U {16..23}; lane>=16 holds row r=lane-16, K in {8..15} U {24..31}.
// B 32x16 mirrors with N<->M roles, so a row-major (N x K) "BT" loads the
// same way. Two contiguous 16B loads per fragment.
static __device__ __forceinline__
v16h load_frag(const _Float16* __restrict__ base, int ld, int row0, int k0, int lane) {
  const int half_id = lane >> 4;
  const int r       = lane & 15;
  const _Float16* p = base + (size_t)(row0 + r) * (size_t)ld + (size_t)k0 + 8 * half_id;
  v8h lo = *(const v8h*)(p);
  v8h hi = *(const v8h*)(p + 16);
  v16h out;
#pragma unroll
  for (int t = 0; t < 8; ++t) { out[t] = lo[t]; out[8 + t] = hi[t]; }
  return out;
}

// ---- prep: f32 -> f16 copies / transposes ---------------------------------
__global__ void prep_kernel(const float* __restrict__ x,
                            const float* __restrict__ w_qkv,
                            const float* __restrict__ w_out,
                            _Float16* __restrict__ xh,
                            _Float16* __restrict__ wqkvT,
                            _Float16* __restrict__ woutT) {
  size_t i = (size_t)blockIdx.x * blockDim.x + threadIdx.x;
  const size_t TX = (size_t)NN_ * DIM_;          // 8388608
  if (i < TX) xh[i] = (_Float16)x[i];
  if (i < (size_t)DIM_ * QKVN_) {                // w_qkv[k][j] -> wqkvT[j][k]
    int k = (int)(i / QKVN_), j = (int)(i % QKVN_);
    wqkvT[(size_t)j * DIM_ + k] = (_Float16)w_qkv[i];
  }
  if (i < (size_t)DIM_ * DIM_) {                 // w_out[k][j] -> woutT[j][k]
    int k = (int)(i / DIM_), j = (int)(i % DIM_);
    woutT[(size_t)j * DIM_ + k] = (_Float16)w_out[i];
  }
}

__global__ void zero_u32(unsigned int* p) { *p = 0u; }

// ---- QKV GEMM: [NN,512] x [512,1536], epilogue routes to q/k/v heads ------
__global__ __launch_bounds__(256)
void qkv_kernel(const _Float16* __restrict__ Xh, const _Float16* __restrict__ WT,
                _Float16* __restrict__ q, _Float16* __restrict__ k,
                _Float16* __restrict__ v, _Float16* __restrict__ vT) {
  const int lane = threadIdx.x & 31;
  const int wave = threadIdx.x >> 5;
  const int tiles_y = QKVN_ / 16;                // 96
  const int tiles   = (NN_ / 16) * tiles_y;      // 98304
  int tile = blockIdx.x * 8 + wave;
  if (tile >= tiles) return;
  const int ti = tile / tiles_y, tj = tile % tiles_y;
  v8f acc = {};
  for (int k0 = 0; k0 < DIM_; k0 += 32) {
    if (k0 + 32 < DIM_) {
      __builtin_prefetch(Xh + (size_t)(ti * 16 + lane) * DIM_ + k0 + 32, 0, 0);
      __builtin_prefetch(WT + (size_t)(tj * 16 + (lane & 15)) * DIM_ + k0 + 32, 0, 0);
    }
    v16h a = load_frag(Xh, DIM_, ti * 16, k0, lane);
    v16h b = load_frag(WT, DIM_, tj * 16, k0, lane);
    acc = __builtin_amdgcn_wmma_f32_16x16x32_f16(false, a, false, b, (short)0, acc, false, false);
  }
  const int cr = 8 * (lane >> 4), cc = lane & 15;
#pragma unroll
  for (int p = 0; p < 8; ++p) {
    const int row = ti * 16 + cr + p;            // b*N + n
    const int col = tj * 16 + cc;                // which*512 + h*64 + d
    const int bb = row >> 12, nn = row & (N_ - 1);
    const int which = col >> 9, hh = (col >> 6) & 7, dd = col & 63;
    const size_t idx = (((size_t)(bb * H_ + hh)) * N_ + nn) * DH_ + dd;
    const float val = acc[p];
    if (which == 0) {
      q[idx] = (_Float16)(val * 0.125f);         // DH^-0.5
    } else if (which == 1) {
      k[idx] = (_Float16)val;
    } else {
      v[idx] = (_Float16)val;
      vT[(((size_t)(bb * H_ + hh)) * DH_ + dd) * N_ + nn] = (_Float16)val;
    }
  }
}

// ---- landmark means -------------------------------------------------------
__global__ void landmark_kernel(const _Float16* __restrict__ q, const _Float16* __restrict__ k,
                                _Float16* __restrict__ ql, _Float16* __restrict__ kl) {
  int i = blockIdx.x * blockDim.x + threadIdx.x;         // BH*M*DH = 524288
  if (i >= BH_ * M_ * DH_) return;
  const int d = i & 63, m = (i >> 6) & (M_ - 1), bh = i >> 14;
  const size_t base = (((size_t)bh) * N_ + (size_t)m * L_) * DH_ + d;
  float sq = 0.f, sk = 0.f;
#pragma unroll
  for (int t = 0; t < L_; ++t) {
    sq += (float)q[base + (size_t)t * DH_];
    sk += (float)k[base + (size_t)t * DH_];
  }
  ql[i] = (_Float16)(sq * (1.f / L_));
  kl[i] = (_Float16)(sk * (1.f / L_));
}

// ---- generic batched GEMM, f32 out: out = A(MxK) @ BT(NxK)^T --------------
__global__ __launch_bounds__(256)
void gemm_f32(const _Float16* __restrict__ A, int lda, long long sA,
              const _Float16* __restrict__ BT, int ldb, long long sB,
              float* __restrict__ out, int ldo, long long sO,
              int Mr, int Nc, int K) {
  const int lane = threadIdx.x & 31;
  const int wave = threadIdx.x >> 5;
  const int tiles_y = Nc >> 4;
  const int tiles   = (Mr >> 4) * tiles_y;
  int tile = blockIdx.x * 8 + wave;
  if (tile >= tiles) return;
  const int ti = tile / tiles_y, tj = tile % tiles_y;
  const _Float16* Ab = A + (size_t)blockIdx.y * sA;
  const _Float16* Bb = BT + (size_t)blockIdx.y * sB;
  float* Ob = out + (size_t)blockIdx.y * sO;
  v8f acc = {};
  for (int k0 = 0; k0 < K; k0 += 32) {
    if (k0 + 32 < K) {
      __builtin_prefetch(Ab + (size_t)(ti * 16 + lane) * lda + k0 + 32, 0, 0);
      __builtin_prefetch(Bb + (size_t)(tj * 16 + (lane & 15)) * ldb + k0 + 32, 0, 0);
    }
    v16h a = load_frag(Ab, lda, ti * 16, k0, lane);
    v16h b = load_frag(Bb, ldb, tj * 16, k0, lane);
    acc = __builtin_amdgcn_wmma_f32_16x16x32_f16(false, a, false, b, (short)0, acc, false, false);
  }
  const int cr = 8 * (lane >> 4), cc = lane & 15;
#pragma unroll
  for (int p = 0; p < 8; ++p)
    Ob[(size_t)(ti * 16 + cr + p) * ldo + tj * 16 + cc] = acc[p];
}

// ---- generic batched GEMM, f16 out (+optional C add, +optional outT) ------
// out = scaleAB * (A @ BT^T) + scaleC * C ; outT = out^T
__global__ __launch_bounds__(256)
void gemm_f16(const _Float16* __restrict__ A, int lda, long long sA,
              const _Float16* __restrict__ BT, int ldb, long long sB,
              const _Float16* __restrict__ Cadd, int ldc, long long sC,
              _Float16* __restrict__ out, int ldo, long long sO,
              _Float16* __restrict__ outT, int ldt, long long sT,
              float scaleAB, float scaleC,
              int Mr, int Nc, int K) {
  const int lane = threadIdx.x & 31;
  const int wave = threadIdx.x >> 5;
  const int tiles_y = Nc >> 4;
  const int tiles   = (Mr >> 4) * tiles_y;
  int tile = blockIdx.x * 8 + wave;
  if (tile >= tiles) return;
  const int ti = tile / tiles_y, tj = tile % tiles_y;
  const _Float16* Ab = A + (size_t)blockIdx.y * sA;
  const _Float16* Bb = BT + (size_t)blockIdx.y * sB;
  v8f acc = {};
  for (int k0 = 0; k0 < K; k0 += 32) {
    if (k0 + 32 < K) {
      __builtin_prefetch(Ab + (size_t)(ti * 16 + lane) * lda + k0 + 32, 0, 0);
      __builtin_prefetch(Bb + (size_t)(tj * 16 + (lane & 15)) * ldb + k0 + 32, 0, 0);
    }
    v16h a = load_frag(Ab, lda, ti * 16, k0, lane);
    v16h b = load_frag(Bb, ldb, tj * 16, k0, lane);
    acc = __builtin_amdgcn_wmma_f32_16x16x32_f16(false, a, false, b, (short)0, acc, false, false);
  }
  const int cr = 8 * (lane >> 4), cc = lane & 15;
  _Float16* Ob = out + (size_t)blockIdx.y * sO;
  _Float16* Tb = outT ? (outT + (size_t)blockIdx.y * sT) : nullptr;
  const _Float16* Cb = Cadd ? (Cadd + (size_t)blockIdx.y * sC) : nullptr;
#pragma unroll
  for (int p = 0; p < 8; ++p) {
    const int row = ti * 16 + cr + p;
    const int col = tj * 16 + cc;
    float val = scaleAB * acc[p];
    if (Cb) val += scaleC * (float)Cb[(size_t)row * ldc + col];
    Ob[(size_t)row * ldo + col] = (_Float16)val;
    if (Tb) Tb[(size_t)col * ldt + row] = (_Float16)val;
  }
}

// ---- row softmax: f32 in -> f16 out ---------------------------------------
__global__ void softmax_rows(const float* __restrict__ in, _Float16* __restrict__ out, int C) {
  __shared__ float red[256];
  const size_t row = blockIdx.x;
  const float* rin = in + row * (size_t)C;
  _Float16* rout = out + row * (size_t)C;
  const int tid = threadIdx.x;
  float mx = -3.4e38f;
  for (int j = tid; j < C; j += 256) mx = fmaxf(mx, rin[j]);
  red[tid] = mx; __syncthreads();
  for (int s = 128; s > 0; s >>= 1) { if (tid < s) red[tid] = fmaxf(red[tid], red[tid + s]); __syncthreads(); }
  mx = red[0]; __syncthreads();
  float sum = 0.f;
  for (int j = tid; j < C; j += 256) sum += __expf(rin[j] - mx);
  red[tid] = sum; __syncthreads();
  for (int s = 128; s > 0; s >>= 1) { if (tid < s) red[tid] += red[tid + s]; __syncthreads(); }
  const float inv = 1.f / red[0];
  for (int j = tid; j < C; j += 256) rout[j] = (_Float16)(__expf(rin[j] - mx) * inv);
}

// ---- global max of column sums of attn2 (row sums are exactly 1) ----------
__global__ void colsum_max(const _Float16* __restrict__ attn2, unsigned int* __restrict__ denomBits) {
  __shared__ float red[256];
  const int bh = blockIdx.x, t = threadIdx.x;
  const _Float16* m = attn2 + (size_t)bh * (M_ * M_);
  float s = 0.f;
  for (int r = 0; r < M_; ++r) s += (float)m[(size_t)r * M_ + t];
  red[t] = s; __syncthreads();
  for (int st = 128; st > 0; st >>= 1) { if (t < st) red[t] = fmaxf(red[t], red[t + st]); __syncthreads(); }
  if (t == 0) atomicMax(denomBits, __float_as_uint(red[0]));   // positive floats: uint order == float order
}

// ---- z0 = attn2^T / denom (write z0 and z0T) ------------------------------
__global__ void z0_kernel(const _Float16* __restrict__ attn2, const unsigned int* __restrict__ denomBits,
                          _Float16* __restrict__ z, _Float16* __restrict__ zT) {
  size_t i = (size_t)blockIdx.x * blockDim.x + threadIdx.x;   // BH*M*M = 2097152
  if (i >= (size_t)BH_ * M_ * M_) return;
  const float inv = 1.f / (__uint_as_float(*denomBits) * 1.0f + 1e-8f); // max(rowsum)==1
  const int j = (int)(i & (M_ - 1)), r = (int)((i >> 8) & (M_ - 1));
  const size_t bh = i >> 16;
  z[i]  = (_Float16)((float)attn2[(bh << 16) + (size_t)j * M_ + r] * inv);
  zT[i] = (_Float16)((float)attn2[i] * inv);
}

// ---- depthwise conv residual (33 taps along n, per head) into [b][n][h][d]
__global__ void conv_res_kernel(const _Float16* __restrict__ v, const float* __restrict__ w_res,
                                _Float16* __restrict__ res) {
  size_t i = (size_t)blockIdx.x * blockDim.x + threadIdx.x;   // B*N*H*DH = 8388608
  if (i >= (size_t)B_ * N_ * H_ * DH_) return;
  const int d = (int)(i & 63), h = (int)((i >> 6) & 7);
  const int n = (int)((i >> 9) & (N_ - 1)), b = (int)(i >> 21);
  const _Float16* vp = v + (((size_t)(b * H_ + h)) * N_) * DH_ + d;
  const float* w = w_res + h * 33;
  float s = 0.f;
#pragma unroll
  for (int t = 0; t < 33; ++t) {
    const int nn = n + t - 16;
    if (nn >= 0 && nn < N_) s += w[t] * (float)vp[(size_t)nn * DH_];
  }
  res[i] = (_Float16)s;
}

// ---- out_pre[b][n][h*64+d] = (attn1 @ Y)[bh][n][d] + res ------------------
__global__ __launch_bounds__(256)
void attn_out_kernel(const _Float16* __restrict__ attn1, const _Float16* __restrict__ YT,
                     const _Float16* __restrict__ res, _Float16* __restrict__ out_pre) {
  const int lane = threadIdx.x & 31;
  const int wave = threadIdx.x >> 5;
  const int tiles = (N_ / 16) * (DH_ / 16);                  // 1024 per bh
  int tile = blockIdx.x * 8 + wave;
  if (tile >= tiles) return;
  const int ti = tile >> 2, tj = tile & 3;
  const int bh = blockIdx.y, b = bh >> 3, h = bh & 7;
  const _Float16* Ab = attn1 + (size_t)bh * N_ * M_;
  const _Float16* Bb = YT + (size_t)bh * DH_ * M_;
  v8f acc = {};
  for (int k0 = 0; k0 < M_; k0 += 32) {
    v16h a = load_frag(Ab, M_, ti * 16, k0, lane);
    v16h bfr = load_frag(Bb, M_, tj * 16, k0, lane);
    acc = __builtin_amdgcn_wmma_f32_16x16x32_f16(false, a, false, bfr, (short)0, acc, false, false);
  }
  const int cr = 8 * (lane >> 4), cc = lane & 15;
#pragma unroll
  for (int p = 0; p < 8; ++p) {
    const int n = ti * 16 + cr + p;
    const int d = tj * 16 + cc;
    const size_t oidx = (((size_t)(b * N_ + n)) * H_ + h) * DH_ + d;
    out_pre[oidx] = (_Float16)(acc[p] + (float)res[oidx]);
  }
}

// ---------------------------------------------------------------------------
extern "C" void kernel_launch(void* const* d_in, const int* in_sizes, int n_in,
                              void* d_out, int out_size, void* d_ws, size_t ws_size,
                              hipStream_t stream) {
  (void)in_sizes; (void)n_in; (void)out_size; (void)ws_size;
  const float* x      = (const float*)d_in[0];
  const float* w_qkv  = (const float*)d_in[1];
  const float* w_out  = (const float*)d_in[2];
  const float* w_res  = (const float*)d_in[3];
  float* out = (float*)d_out;

  // ---- workspace carve-out (deterministic, 256B aligned) ----
  char* ws = (char*)d_ws;
  size_t off = 0;
  auto take = [&](size_t bytes) -> void* {
    void* p = ws + off;
    off = (off + bytes + 255) & ~(size_t)255;
    return p;
  };
  const size_t HB = sizeof(_Float16);
  unsigned int* denom = (unsigned int*)take(4);
  _Float16* XH     = (_Float16*)take((size_t)NN_ * DIM_ * HB);
  _Float16* WQKVT  = (_Float16*)take((size_t)QKVN_ * DIM_ * HB);
  _Float16* WOUTT  = (_Float16*)take((size_t)DIM_ * DIM_ * HB);
  _Float16* QH     = (_Float16*)take((size_t)BH_ * N_ * DH_ * HB);
  _Float16* KH     = (_Float16*)take((size_t)BH_ * N_ * DH_ * HB);
  _Float16* VH     = (_Float16*)take((size_t)BH_ * N_ * DH_ * HB);
  _Float16* VT     = (_Float16*)take((size_t)BH_ * DH_ * N_ * HB);
  _Float16* QL     = (_Float16*)take((size_t)BH_ * M_ * DH_ * HB);
  _Float16* KL     = (_Float16*)take((size_t)BH_ * M_ * DH_ * HB);
  _Float16* ATTN2  = (_Float16*)take((size_t)BH_ * M_ * M_ * HB);
  const size_t MM = (size_t)BH_ * M_ * M_;
  _Float16* Z0  = (_Float16*)take(MM * HB);
  _Float16* Z0T = (_Float16*)take(MM * HB);
  _Float16* Z1  = (_Float16*)take(MM * HB);
  _Float16* Z1T = (_Float16*)take(MM * HB);
  _Float16* XZ  = (_Float16*)take(MM * HB);
  _Float16* XZT = (_Float16*)take(MM * HB);
  _Float16* T1  = (_Float16*)take(MM * HB);
  _Float16* T1T = (_Float16*)take(MM * HB);
  _Float16* T3  = (_Float16*)take(MM * HB);
  _Float16* T3T = (_Float16*)take(MM * HB);
  _Float16* Wb  = (_Float16*)take((size_t)BH_ * M_ * DH_ * HB);
  _Float16* WbT = (_Float16*)take((size_t)BH_ * DH_ * M_ * HB);
  _Float16* Yb  = (_Float16*)take((size_t)BH_ * M_ * DH_ * HB);
  _Float16* YbT = (_Float16*)take((size_t)BH_ * DH_ * M_ * HB);
  _Float16* ATTN1 = (_Float16*)take((size_t)BH_ * N_ * M_ * HB);
  _Float16* ATTN3 = (_Float16*)take((size_t)BH_ * M_ * N_ * HB);
  _Float16* RES   = (_Float16*)take((size_t)B_ * N_ * H_ * DH_ * HB);
  _Float16* OUTPRE= (_Float16*)take((size_t)B_ * N_ * H_ * DH_ * HB);
  float* SIMBIG   = (float*)take((size_t)BH_ * N_ * M_ * sizeof(float)); // reused: sim2, sim1, sim3

  // ---- pipeline ----
  zero_u32<<<1, 1, 0, stream>>>(denom);

  prep_kernel<<<(NN_ * DIM_ + 255) / 256, 256, 0, stream>>>(x, w_qkv, w_out, XH, WQKVT, WOUTT);

  qkv_kernel<<<dim3((NN_ / 16) * (QKVN_ / 16) / 8, 1), 256, 0, stream>>>(XH, WQKVT, QH, KH, VH, VT);

  landmark_kernel<<<(BH_ * M_ * DH_ + 255) / 256, 256, 0, stream>>>(QH, KH, QL, KL);

  // sim2 = q_land @ k_land^T   [BH, 256, 256]
  gemm_f32<<<dim3(((M_ / 16) * (M_ / 16)) / 8, BH_), 256, 0, stream>>>(
      QL, DH_, (long long)M_ * DH_, KL, DH_, (long long)M_ * DH_,
      SIMBIG, M_, (long long)M_ * M_, M_, M_, DH_);
  softmax_rows<<<BH_ * M_, 256, 0, stream>>>(SIMBIG, ATTN2, M_);

  colsum_max<<<BH_, 256, 0, stream>>>(ATTN2, denom);
  z0_kernel<<<(unsigned)((MM + 255) / 256), 256, 0, stream>>>(ATTN2, denom, Z0, Z0T);

  // Moore-Penrose Newton iterations: z' = 3.25 z - 0.25 z@T3 ;
  // T3 = 15 XZ - XZ@T1 ; T1 = 7 XZ - XZ@XZ ; XZ = x@z
  {
    _Float16 *za = Z0, *zaT = Z0T, *zb = Z1, *zbT = Z1T;
    const long long sm = (long long)M_ * M_;
    const dim3 g(((M_ / 16) * (M_ / 16)) / 8, BH_);
    for (int it = 0; it < 6; ++it) {
      gemm_f16<<<g, 256, 0, stream>>>(ATTN2, M_, sm, zaT, M_, sm,
                                      (const _Float16*)nullptr, 0, 0,
                                      XZ, M_, sm, XZT, M_, sm, 1.f, 0.f, M_, M_, M_);
      gemm_f16<<<g, 256, 0, stream>>>(XZ, M_, sm, XZT, M_, sm, XZ, M_, sm,
                                      T1, M_, sm, T1T, M_, sm, -1.f, 7.f, M_, M_, M_);
      gemm_f16<<<g, 256, 0, stream>>>(XZ, M_, sm, T1T, M_, sm, XZ, M_, sm,
                                      T3, M_, sm, T3T, M_, sm, -1.f, 15.f, M_, M_, M_);
      gemm_f16<<<g, 256, 0, stream>>>(za, M_, sm, T3T, M_, sm, za, M_, sm,
                                      zb, M_, sm, zbT, M_, sm, -0.25f, 3.25f, M_, M_, M_);
      _Float16* t;
      t = za; za = zb; zb = t;
      t = zaT; zaT = zbT; zbT = t;
    }
    // za now holds attn2_inv (row-major f16, used as A below)

    // sim1 = q @ k_land^T   [BH, 4096, 256] -> attn1
    gemm_f32<<<dim3(((N_ / 16) * (M_ / 16)) / 8, BH_), 256, 0, stream>>>(
        QH, DH_, (long long)N_ * DH_, KL, DH_, (long long)M_ * DH_,
        SIMBIG, M_, (long long)N_ * M_, N_, M_, DH_);
    softmax_rows<<<BH_ * N_, 256, 0, stream>>>(SIMBIG, ATTN1, M_);

    // sim3 = q_land @ k^T   [BH, 256, 4096] -> attn3 (reuse SIMBIG)
    gemm_f32<<<dim3(((M_ / 16) * (N_ / 16)) / 8, BH_), 256, 0, stream>>>(
        QL, DH_, (long long)M_ * DH_, KH, DH_, (long long)N_ * DH_,
        SIMBIG, N_, (long long)M_ * N_, M_, N_, DH_);
    softmax_rows<<<BH_ * M_, 256, 0, stream>>>(SIMBIG, ATTN3, N_);

    // W = attn3 @ v   [BH, 256, 64]  (B operand via vT, K = 4096)
    gemm_f16<<<dim3(((M_ / 16) * (DH_ / 16)) / 8, BH_), 256, 0, stream>>>(
        ATTN3, N_, (long long)M_ * N_, VT, N_, (long long)DH_ * N_,
        (const _Float16*)nullptr, 0, 0,
        Wb, DH_, (long long)M_ * DH_, WbT, M_, (long long)DH_ * M_, 1.f, 0.f, M_, DH_, N_);

    // Y = attn2_inv @ W   [BH, 256, 64]
    gemm_f16<<<dim3(((M_ / 16) * (DH_ / 16)) / 8, BH_), 256, 0, stream>>>(
        za, M_, sm, WbT, M_, (long long)DH_ * M_,
        (const _Float16*)nullptr, 0, 0,
        Yb, DH_, (long long)M_ * DH_, YbT, M_, (long long)DH_ * M_, 1.f, 0.f, M_, DH_, M_);
  }

  // depthwise conv residual of v into [b][n][h][d]
  conv_res_kernel<<<(B_ * N_ * H_ * DH_ + 255) / 256, 256, 0, stream>>>(VH, w_res, RES);

  // out_pre = attn1 @ Y + res, laid out [b][n][inner]
  attn_out_kernel<<<dim3(((N_ / 16) * (DH_ / 16)) / 8, BH_), 256, 0, stream>>>(ATTN1, YbT, RES, OUTPRE);

  // final projection: [NN, 512] @ [512, 512] -> d_out (f32)
  gemm_f32<<<dim3(((NN_ / 16) * (DIM_ / 16)) / 8, 1), 256, 0, stream>>>(
      OUTPRE, DIM_, 0, WOUTT, DIM_, 0, out, DIM_, 0, NN_, DIM_, DIM_);
}
